// KANLayer_8280696947314
// MI455X (gfx1250) — matrix operand order repaired and checked by reading, not compiled, not run on hardware
//
#include <hip/hip_runtime.h>

// ---------------------------------------------------------------------------
// KAN layer on gfx1250.
// Fast path (needs ~226.5MB workspace):
//   1) prep_weights : Wc(2048,18432) bf16 = [base_w | 0.1*spline_w]
//   2) prep_acts    : Ac(4096,18432) bf16 = [x | relu(x - knot_g) expansion]
//   3) kan_gemm_bf16: out = Ac @ Wc^T -- bf16 WMMA GEMM; LDS tiles filled with
//      GLOBAL_LOAD_ASYNC_TO_LDS_B128 (ASYNCcnt), double-buffered, no VGPR
//      staging and no conversion VALU in the inner loop.
// Fallback: fused kernel (on-the-fly conversion) if ws is too small.
// ---------------------------------------------------------------------------

typedef __attribute__((ext_vector_type(16))) __bf16 v16bf;
typedef __attribute__((ext_vector_type(8)))  float  v8f;

#define BATCH   4096
#define IN_F    2048
#define OUT_F   2048
#define GRID_G  8
#define KSPL    (IN_F * GRID_G)     // 16384
#define KT      (IN_F + KSPL)       // 18432 total fused K
#define SCALE_S 0.1f

#define BM 128
#define BN 128

__device__ __forceinline__ unsigned short f32_to_bf16(float f) {
    unsigned int u = __float_as_uint(f);
    u += 0x7FFFu + ((u >> 16) & 1u);           // round-to-nearest-even
    return (unsigned short)(u >> 16);
}
__device__ __forceinline__ unsigned int pack2(float lo, float hi) {
    return (unsigned int)f32_to_bf16(lo) | ((unsigned int)f32_to_bf16(hi) << 16);
}

union FragU { v16bf v; uint4 q[2]; };

// One 32-deep MMA step over staged LDS tiles: 2 A-frags x 4 B-frags.
template <int STR>
__device__ __forceinline__ void mma_step(const unsigned short* sA,
                                         const unsigned short* sB,
                                         int wm, int wn, int hl, int hi,
                                         v8f acc[2][4]) {
    v16bf afr[2];
    v16bf bfr[4];
#pragma unroll
    for (int t = 0; t < 2; ++t) {
        const unsigned short* ap = &sA[(unsigned)(wm + t * 16 + hl) * STR + hi * 8];
        FragU a;
        a.q[0] = *(const uint4*)(ap);
        a.q[1] = *(const uint4*)(ap + 16);
        afr[t] = a.v;
    }
#pragma unroll
    for (int u = 0; u < 4; ++u) {
        const unsigned short* bp = &sB[(unsigned)(wn + u * 16 + hl) * STR + hi * 8];
        FragU b;
        b.q[0] = *(const uint4*)(bp);
        b.q[1] = *(const uint4*)(bp + 16);
        bfr[u] = b.v;
    }
#pragma unroll
    for (int t = 0; t < 2; ++t)
#pragma unroll
        for (int u = 0; u < 4; ++u)
            acc[t][u] = __builtin_amdgcn_wmma_f32_16x16x32_bf16(
                false, afr[t], false, bfr[u], (short)0, acc[t][u], false, false);
}

__device__ __forceinline__ void store_tile(float* __restrict__ out, v8f acc[2][4],
                                           int row0, int col0, int wm, int wn,
                                           int hl, int hi) {
#pragma unroll
    for (int t = 0; t < 2; ++t) {
#pragma unroll
        for (int u = 0; u < 4; ++u) {
            const int n = col0 + wn + u * 16 + hl;
            const int mbase = row0 + wm + t * 16 + 8 * hi;
#pragma unroll
            for (int r = 0; r < 8; ++r)
                out[(size_t)(mbase + r) * OUT_F + n] = acc[t][u][r];
        }
    }
}

// Async global->LDS copy of one 64B row segment (4 x b128).
// INST_OFFSET is added to both the LDS and the global address (ISA 15.18.3),
// which matches our contiguous segments.
__device__ __forceinline__ void async_copy_row64(unsigned lds, const unsigned short* g) {
    asm volatile("global_load_async_to_lds_b128 %0, %1, off"
                 :: "v"(lds), "v"(g) : "memory");
    asm volatile("global_load_async_to_lds_b128 %0, %1, off offset:16"
                 :: "v"(lds), "v"(g) : "memory");
    asm volatile("global_load_async_to_lds_b128 %0, %1, off offset:32"
                 :: "v"(lds), "v"(g) : "memory");
    asm volatile("global_load_async_to_lds_b128 %0, %1, off offset:48"
                 :: "v"(lds), "v"(g) : "memory");
}
__device__ __forceinline__ void wait_async0() {
    asm volatile("s_wait_asynccnt 0x0" ::: "memory");
}

// =================== Fast path: prep kernels ===================

__global__ __launch_bounds__(256)
void prep_weights(const float* __restrict__ bw, const float* __restrict__ sw,
                  unsigned short* __restrict__ Wc) {
    const size_t base = ((size_t)blockIdx.x * blockDim.x + threadIdx.x) * 8;
    const int row = (int)(base / KT);
    const int col = (int)(base % KT);
    const float* src;
    float s;
    if (col < IN_F) { src = bw + (size_t)row * IN_F + col;            s = 1.0f; }
    else            { src = sw + (size_t)row * KSPL + (col - IN_F);   s = SCALE_S; }
    float4 a = ((const float4*)src)[0];
    float4 b = ((const float4*)src)[1];
    uint4 q;
    q.x = pack2(s * a.x, s * a.y);
    q.y = pack2(s * a.z, s * a.w);
    q.z = pack2(s * b.x, s * b.y);
    q.w = pack2(s * b.z, s * b.w);
    *(uint4*)(Wc + base) = q;
}

__global__ __launch_bounds__(256)
void prep_acts(const float* __restrict__ x, const float* __restrict__ knots,
               unsigned short* __restrict__ Ac) {
    const size_t base = ((size_t)blockIdx.x * blockDim.x + threadIdx.x) * 8;
    const int row = (int)(base / KT);
    const int col = (int)(base % KT);
    uint4 q;
    if (col < IN_F) {
        const float* src = x + (size_t)row * IN_F + col;
        float4 a = ((const float4*)src)[0];
        float4 b = ((const float4*)src)[1];
        q.x = pack2(a.x, a.y); q.y = pack2(a.z, a.w);
        q.z = pack2(b.x, b.y); q.w = pack2(b.z, b.w);
    } else {
        const int i = (col - IN_F) >> 3;
        const float v = x[(size_t)row * IN_F + i];
        q.x = pack2(fmaxf(v - knots[0], 0.f), fmaxf(v - knots[1], 0.f));
        q.y = pack2(fmaxf(v - knots[2], 0.f), fmaxf(v - knots[3], 0.f));
        q.z = pack2(fmaxf(v - knots[4], 0.f), fmaxf(v - knots[5], 0.f));
        q.w = pack2(fmaxf(v - knots[6], 0.f), fmaxf(v - knots[7], 0.f));
    }
    *(uint4*)(Ac + base) = q;
}

// =================== Fast path: pure bf16 GEMM ===================

#define KC2   64     // K per stage (two 32-deep WMMA steps)
#define SSTR2 72     // LDS row stride in halves: 144B, 16B-aligned

__global__ __launch_bounds__(256)
void kan_gemm_bf16(const unsigned short* __restrict__ Ac,
                   const unsigned short* __restrict__ Wc,
                   float* __restrict__ out) {
    __shared__ unsigned short sA[2][BM * SSTR2];   // 2 x 18KB
    __shared__ unsigned short sB[2][BN * SSTR2];   // 2 x 18KB

    const int tid  = threadIdx.x;
    const int lane = tid & 31;
    const int wid  = tid >> 5;
    const int wm   = (wid & 3) * 32;
    const int wn   = (wid >> 2) * 64;
    const int hl   = lane & 15;
    const int hi   = lane >> 4;

    const int nTilesN = OUT_F / BN;
    const int row0 = (blockIdx.x / nTilesN) * BM;
    const int col0 = (blockIdx.x % nTilesN) * BN;

    v8f zero = {};
    v8f acc[2][4];
#pragma unroll
    for (int t = 0; t < 2; ++t)
#pragma unroll
        for (int u = 0; u < 4; ++u) acc[t][u] = zero;

    // Loader mapping: 2 threads per row, 32 contiguous halves (64B) each.
    const int lr = tid >> 1;
    const int lc = (tid & 1) * 32;

    const unsigned short* gA = Ac + (size_t)(row0 + lr) * KT + lc;
    const unsigned short* gB = Wc + (size_t)(col0 + lr) * KT + lc;

    // LDS byte addresses (low 32 bits of the generic pointer = DS offset).
    unsigned ldsA[2], ldsB[2];
    ldsA[0] = (unsigned)(size_t)&sA[0][(unsigned)lr * SSTR2 + lc];
    ldsA[1] = (unsigned)(size_t)&sA[1][(unsigned)lr * SSTR2 + lc];
    ldsB[0] = (unsigned)(size_t)&sB[0][(unsigned)lr * SSTR2 + lc];
    ldsB[1] = (unsigned)(size_t)&sB[1][(unsigned)lr * SSTR2 + lc];

    // Prologue: chunk 0 -> buffer 0 (async, no VGPR staging).
    async_copy_row64(ldsA[0], gA);
    async_copy_row64(ldsB[0], gB);

    const int NCH = KT / KC2;              // 288 stages
    for (int ch = 0; ch < NCH; ++ch) {
        wait_async0();                     // my copies for chunk ch landed
        __syncthreads();                   // everyone landed; prior readers done
        if (ch + 1 < NCH) {                // overlap next copy with this MMA
            const int nb = (ch + 1) & 1;
            async_copy_row64(ldsA[nb], gA + (size_t)(ch + 1) * KC2);
            async_copy_row64(ldsB[nb], gB + (size_t)(ch + 1) * KC2);
        }
        const unsigned short* rA = &sA[ch & 1][0];
        const unsigned short* rB = &sB[ch & 1][0];
        mma_step<SSTR2>(rA,      rB,      wm, wn, hl, hi, acc);
        mma_step<SSTR2>(rA + 32, rB + 32, wm, wn, hl, hi, acc);
    }

    store_tile(out, acc, row0, col0, wm, wn, hl, hi);
}

// =================== Fallback: fused kernel ===================

#define KC   32
#define SSTR 40

__global__ __launch_bounds__(256)
void kan_fused_wmma(const float* __restrict__ x,
                    const float* __restrict__ bw,
                    const float* __restrict__ sw,
                    const float* __restrict__ knots,
                    float* __restrict__ out) {
    __shared__ unsigned short sA[BM * SSTR];
    __shared__ unsigned short sB[BN * SSTR];

    const int tid  = threadIdx.x;
    const int lane = tid & 31;
    const int wid  = tid >> 5;
    const int wm   = (wid & 3) * 32;
    const int wn   = (wid >> 2) * 64;
    const int hl   = lane & 15;
    const int hi   = lane >> 4;

    const int nTilesN = OUT_F / BN;
    const int row0 = (blockIdx.x / nTilesN) * BM;
    const int col0 = (blockIdx.x % nTilesN) * BN;

    float kn[GRID_G];
#pragma unroll
    for (int g = 0; g < GRID_G; ++g) kn[g] = knots[g];

    v8f zero = {};
    v8f acc[2][4];
#pragma unroll
    for (int t = 0; t < 2; ++t)
#pragma unroll
        for (int u = 0; u < 4; ++u) acc[t][u] = zero;

    const int lr = tid >> 1;
    const int lc = (tid & 1) * 16;

    for (int kc = 0; kc < IN_F; kc += KC) {
        const float* xp = x  + (size_t)(row0 + lr) * IN_F + kc + lc;
        const float* wp = bw + (size_t)(col0 + lr) * IN_F + kc + lc;
        __builtin_prefetch((const char*)wp + KC * sizeof(float), 0, 1);
        float4 x0 = ((const float4*)xp)[0], x1 = ((const float4*)xp)[1];
        float4 x2 = ((const float4*)xp)[2], x3 = ((const float4*)xp)[3];
        float4 w0 = ((const float4*)wp)[0], w1 = ((const float4*)wp)[1];
        float4 w2 = ((const float4*)wp)[2], w3 = ((const float4*)wp)[3];

        __syncthreads();
        uint4 qa, qb;
        qa.x = pack2(x0.x, x0.y); qa.y = pack2(x0.z, x0.w);
        qa.z = pack2(x1.x, x1.y); qa.w = pack2(x1.z, x1.w);
        qb.x = pack2(x2.x, x2.y); qb.y = pack2(x2.z, x2.w);
        qb.z = pack2(x3.x, x3.y); qb.w = pack2(x3.z, x3.w);
        *(uint4*)&sA[(unsigned)lr * SSTR + lc]     = qa;
        *(uint4*)&sA[(unsigned)lr * SSTR + lc + 8] = qb;
        qa.x = pack2(w0.x, w0.y); qa.y = pack2(w0.z, w0.w);
        qa.z = pack2(w1.x, w1.y); qa.w = pack2(w1.z, w1.w);
        qb.x = pack2(w2.x, w2.y); qb.y = pack2(w2.z, w2.w);
        qb.z = pack2(w3.x, w3.y); qb.w = pack2(w3.z, w3.w);
        *(uint4*)&sB[(unsigned)lr * SSTR + lc]     = qa;
        *(uint4*)&sB[(unsigned)lr * SSTR + lc + 8] = qb;
        __syncthreads();

        mma_step<SSTR>(sA, sB, wm, wn, hl, hi, acc);
    }

    for (int kc = 0; kc < KSPL; kc += KC) {
        const int ic = kc >> 3;
        const float* wp = sw + (size_t)(col0 + lr) * KSPL + kc + lc;
        __builtin_prefetch((const char*)wp + KC * sizeof(float), 0, 1);
        float4 w0 = ((const float4*)wp)[0], w1 = ((const float4*)wp)[1];
        float4 w2 = ((const float4*)wp)[2], w3 = ((const float4*)wp)[3];
        const float* xp = x + (size_t)(row0 + lr) * IN_F + ic + (tid & 1) * 2;
        float2 xv = *(const float2*)xp;

        __syncthreads();
        uint4 qa, qb;
        qa.x = pack2(SCALE_S * w0.x, SCALE_S * w0.y);
        qa.y = pack2(SCALE_S * w0.z, SCALE_S * w0.w);
        qa.z = pack2(SCALE_S * w1.x, SCALE_S * w1.y);
        qa.w = pack2(SCALE_S * w1.z, SCALE_S * w1.w);
        qb.x = pack2(SCALE_S * w2.x, SCALE_S * w2.y);
        qb.y = pack2(SCALE_S * w2.z, SCALE_S * w2.w);
        qb.z = pack2(SCALE_S * w3.x, SCALE_S * w3.y);
        qb.w = pack2(SCALE_S * w3.z, SCALE_S * w3.w);
        *(uint4*)&sB[(unsigned)lr * SSTR + lc]     = qa;
        *(uint4*)&sB[(unsigned)lr * SSTR + lc + 8] = qb;

        uint4 e0, e1;
        e0.x = pack2(fmaxf(xv.x - kn[0], 0.f), fmaxf(xv.x - kn[1], 0.f));
        e0.y = pack2(fmaxf(xv.x - kn[2], 0.f), fmaxf(xv.x - kn[3], 0.f));
        e0.z = pack2(fmaxf(xv.x - kn[4], 0.f), fmaxf(xv.x - kn[5], 0.f));
        e0.w = pack2(fmaxf(xv.x - kn[6], 0.f), fmaxf(xv.x - kn[7], 0.f));
        e1.x = pack2(fmaxf(xv.y - kn[0], 0.f), fmaxf(xv.y - kn[1], 0.f));
        e1.y = pack2(fmaxf(xv.y - kn[2], 0.f), fmaxf(xv.y - kn[3], 0.f));
        e1.z = pack2(fmaxf(xv.y - kn[4], 0.f), fmaxf(xv.y - kn[5], 0.f));
        e1.w = pack2(fmaxf(xv.y - kn[6], 0.f), fmaxf(xv.y - kn[7], 0.f));
        *(uint4*)&sA[(unsigned)lr * SSTR + lc]     = e0;
        *(uint4*)&sA[(unsigned)lr * SSTR + lc + 8] = e1;
        __syncthreads();

        mma_step<SSTR>(sA, sB, wm, wn, hl, hi, acc);
    }

    store_tile(out, acc, row0, col0, wm, wn, hl, hi);
}

// =================== Launch ===================

extern "C" void kernel_launch(void* const* d_in, const int* in_sizes, int n_in,
                              void* d_out, int out_size, void* d_ws, size_t ws_size,
                              hipStream_t stream) {
    (void)in_sizes; (void)n_in; (void)out_size;
    const float* x     = (const float*)d_in[0];
    const float* bw    = (const float*)d_in[1];
    const float* sw    = (const float*)d_in[2];
    const float* knots = (const float*)d_in[3];
    float* out = (float*)d_out;

    const int grid = (BATCH / BM) * (OUT_F / BN);      // 512 workgroups

    const size_t needW = (size_t)OUT_F * KT * sizeof(unsigned short);  // 75.5MB
    const size_t needA = (size_t)BATCH * KT * sizeof(unsigned short);  // 151MB

    if (ws_size >= needW + needA) {
        unsigned short* Wc = (unsigned short*)d_ws;
        unsigned short* Ac = Wc + (size_t)OUT_F * KT;
        const int gw_blocks = (int)(((size_t)OUT_F * KT / 8) / 256);   // 18432
        const int ga_blocks = (int)(((size_t)BATCH * KT / 8) / 256);   // 36864
        prep_weights<<<gw_blocks, 256, 0, stream>>>(bw, sw, Wc);
        prep_acts   <<<ga_blocks, 256, 0, stream>>>(x, knots, Ac);
        kan_gemm_bf16<<<grid, 256, 0, stream>>>(Ac, Wc, out);
    } else {
        kan_fused_wmma<<<grid, 256, 0, stream>>>(x, bw, sw, knots, out);
    }
}
